// SphericalSpectralTimeConv_30751965839468
// MI455X (gfx1250) — matrix operand order repaired and checked by reading, not compiled, not run on hardware
//
#include <hip/hip_runtime.h>
#include <hip/hip_bf16.h>

typedef __attribute__((ext_vector_type(16))) _Float16 v16h;
typedef __attribute__((ext_vector_type(8)))  float    v8f;
typedef _Float16 h16;

#define DPI 3.14159265358979323846

// ---------------- Wigner-d basis (f64 recursion, f16 output) ----------------
// LEGf[mi][l][t] = -(Nl * d) * sin(th)*pi/128   (forward, with quadrature weight)
// LEGi[mi][t][l] = -(Nl * d)                    (inverse)
// mi in [0,127], window m = mi-63 ; mi==127 pad slice -> 0 ; l in [0,64)
__global__ void k_basis(h16* __restrict__ LEGf, h16* __restrict__ LEGi) {
  int t  = threadIdx.x;   // 0..127
  int mi = blockIdx.x;    // 0..127
  if (mi == 127) {
    for (int l = 0; l < 64; ++l) {
      LEGf[(mi * 64 + l) * 128 + t] = (h16)0.0f;
      LEGi[(mi * 128 + t) * 64 + l] = (h16)0.0f;
    }
    return;
  }
  int m = mi - 63;
  int am = m < 0 ? -m : m;
  double theta = (2.0 * t + 1.0) * DPI / 255.0;
  double c  = cos(theta);
  double c2 = cos(theta * 0.5), s2 = sin(theta * 0.5);
  double lc2 = log(fmax(c2, 1e-300)), ls2 = log(fmax(s2, 1e-300));
  double w = sin(theta) * DPI / 128.0;
  double dj = 0.0, djm1 = 0.0;
  LEGf[(mi * 64 + 0) * 128 + t] = (h16)0.0f;   // l = 0 row is zero (spin-1)
  LEGi[(mi * 128 + t) * 64 + 0] = (h16)0.0f;
  for (int l = 1; l < 64; ++l) {
    double val;
    if (am > l) {
      val = 0.0;
    } else if (m == l) {       // n = -1 : (-1)^(l+1) * exp(binln(2l,l+1) + (l-1)lc2 + (l+1)ls2)
      double b = 0.5 * (lgamma(2.0 * l + 1.0) - lgamma((double)l + 2.0) - lgamma((double)l));
      val = ((l & 1) ? 1.0 : -1.0) * exp(b + (l - 1) * lc2 + (l + 1) * ls2);
    } else if (m == -l) {      // exp(binln(2l,l-1) + (l+1)lc2 + (l-1)ls2)
      double b = 0.5 * (lgamma(2.0 * l + 1.0) - lgamma((double)l) - lgamma((double)l + 2.0));
      val = exp(b + (l + 1) * lc2 + (l - 1) * ls2);
    } else if (l == 1) {       // only m == 0 here
      val = -sin(theta) / sqrt(2.0);
    } else {                   // three-term recursion in l  (n = -1)
      double j   = (double)(l - 1);
      double aj  = sqrt((j * j - (double)m * m) * (j * j - 1.0));
      double aj1 = sqrt(((double)l * l - (double)m * m) * ((double)l * l - 1.0));
      val = ((2.0 * j + 1.0) * (j * (j + 1.0) * c + (double)m) * dj - (j + 1.0) * aj * djm1)
            / (j * aj1);
    }
    djm1 = dj; dj = val;
    double Nl = sqrt((2.0 * l + 1.0) / (4.0 * DPI));
    LEGf[(mi * 64 + l) * 128 + t] = (h16)(-(Nl * val) * w);
    LEGi[(mi * 128 + t) * 64 + l] = (h16)(-(Nl * val));
  }
}

// ---------------- Fourier bases ----------------
// FFT: 2 planes [128 mi][256 p] : re = cos(m*ph)*s , im = -sin(m*ph)*s , s = 2pi/255
// A2 : [256 p][256 k] : k<128 -> cos(m*ph) (mi=k) ; k>=128 -> -sin(m*ph) (mi=k-128)
__global__ void k_fourier(h16* __restrict__ FFT, h16* __restrict__ A2) {
  int col = threadIdx.x;   // 0..255
  int row = blockIdx.x;    // 0..255
  if (row < 128) {
    h16 fr = (h16)0.0f, fi = (h16)0.0f;
    if (row < 127 && col < 255) {
      double m  = (double)(row - 63);
      double ph = 2.0 * DPI * (double)col / 255.0;
      double s  = 2.0 * DPI / 255.0;
      fr = (h16)(cos(ph * m) * s);
      fi = (h16)(-sin(ph * m) * s);
    }
    FFT[row * 256 + col]             = fr;
    FFT[128 * 256 + row * 256 + col] = fi;
  }
  h16 a = (h16)0.0f;
  int mi2 = col & 127;
  if (row < 255 && mi2 < 127) {
    double m  = (double)(mi2 - 63);
    double ph = 2.0 * DPI * (double)row / 255.0;
    a = (col < 128) ? (h16)cos(m * ph) : (h16)(-sin(m * ph));
  }
  A2[row * 256 + col] = a;
}

// ---------------- time embedding -> complex scalar per l ----------------
__global__ void k_tc(const float* __restrict__ t_emb,
                     const float* __restrict__ drw, const float* __restrict__ drb,
                     const float* __restrict__ diw, const float* __restrict__ dib,
                     float* __restrict__ tc) {
  int l = threadIdx.x;   // 0..63
  float sr = drb[l], si = dib[l];
  for (int k = 0; k < 256; ++k) {
    float e = t_emb[k];
    sr += e * drw[k * 64 + l];
    si += e * diw[k * 64 + l];
  }
  tc[l]      = sr;
  tc[64 + l] = si;
}

// -------- BbigT[l] : transposed block matrix [[Ar,Ai],[-Ai,Ar]]^T (256x256 f16) --------
// Logical B[k][n] ; stored BT[n][k]. Ar = wr*tr - wi*ti + conv_kernel ; Ai = wr*ti + wi*tr
__global__ void k_build_bbig(const float* __restrict__ wr_, const float* __restrict__ wi_,
                             const float* __restrict__ ck, const float* __restrict__ tc,
                             h16* __restrict__ BbigT) {
  int idx = blockIdx.x * blockDim.x + threadIdx.x;  // 64*128*128
  int l = idx >> 14;
  int i = (idx >> 7) & 127;
  int o = idx & 127;
  float tr = tc[l], ti = tc[64 + l];
  float wr = wr_[idx], wi = wi_[idx];
  float Ar = wr * tr - wi * ti + ck[i * 128 + o];
  float Ai = wr * ti + wi * tr;
  h16* B = BbigT + (long)l * 65536;
  B[o * 256 + i]               = (h16)Ar;    // B[i][o]       = Ar
  B[(128 + o) * 256 + i]       = (h16)Ai;    // B[i][128+o]   = Ai
  B[o * 256 + 128 + i]         = (h16)(-Ai); // B[128+i][o]   = -Ai
  B[(128 + o) * 256 + 128 + i] = (h16)Ar;    // B[128+i][128+o] = Ar
}

// ------------- repack kernels (f32 -> f16, B operands stored N-major/K-contiguous) -------------
__global__ void k_pack_xh(const float* __restrict__ x, h16* __restrict__ xhT) {
  int idx = blockIdx.x * blockDim.x + threadIdx.x;   // 128*128*256 : [t][c][p]
  int t = idx >> 15;
  int c = (idx >> 8) & 127;
  int p = idx & 255;
  xhT[idx] = (p < 255) ? (h16)x[(t * 255 + p) * 128 + c] : (h16)0.0f;
}

__global__ void k_pack_FmB(const float* __restrict__ Fm32, h16* __restrict__ FmBT) {
  int idx = blockIdx.x * blockDim.x + threadIdx.x;   // 128*256*128 : [m][k][t]
  int mi = idx >> 15;
  int k  = (idx >> 7) & 255;
  int t  = idx & 127;
  FmBT[idx] = (h16)Fm32[t * 32768 + (k >> 7) * 16384 + mi * 128 + (k & 127)];
}

__global__ void k_pack_flmA(const float* __restrict__ flm32, h16* __restrict__ flmA) {
  int idx = blockIdx.x * blockDim.x + threadIdx.x;   // 64*128*256 : [l][m][k]  (A, row-major)
  int l = idx >> 15;
  int m = (idx >> 8) & 127;
  int k = idx & 255;
  flmA[idx] = (h16)flm32[m * 16384 + l * 256 + k];
}

__global__ void k_pack_yB(const float* __restrict__ y32, const float* __restrict__ bias,
                          h16* __restrict__ yBT) {
  int idx = blockIdx.x * blockDim.x + threadIdx.x;   // 128*256*64 : [m][n][l]
  int m = idx >> 14;
  int n = (idx >> 6) & 255;
  int l = idx & 63;
  float v = y32[l * 32768 + m * 256 + n];
  if (n < 128) v += bias[n];                         // conv bias enters y_r only
  yBT[idx] = (h16)v;
}

__global__ void k_pack_GB(const float* __restrict__ G32, h16* __restrict__ GBT) {
  int idx = blockIdx.x * blockDim.x + threadIdx.x;   // 128*128*256 : [t][c][k]
  int t = idx >> 15;
  int c = (idx >> 8) & 127;
  int k = idx & 255;
  float v = (k < 128) ? G32[k * 32768 + t * 256 + c]
                      : G32[(k - 128) * 32768 + t * 256 + 128 + c];
  GBT[idx] = (h16)v;
}

// ---------------- generic batched WMMA GEMM : C(f32) = A(f16) * B(f16) ----------------
// grid = (N/64, M/16, batch), block = 32 (one wave computes a 16x64 strip, 4 accumulators)
// A row-major MxK (lda) ; BT = B transposed, N-major NxK (ldbT) ; C row-major MxN (ldc).
// K % 32 == 0. All fragment loads are K-contiguous -> global_load_b128.
__global__ void k_gemm_wmma(const h16* __restrict__ A, const h16* __restrict__ BT,
                            float* __restrict__ C, int K, int lda, int ldbT, int ldc,
                            long sA, long sB, long sC) {
  const int lane = threadIdx.x;
  const int r    = lane & 15;
  const int half = lane >> 4;
  const h16* Ab = A  + (long)blockIdx.z * sA;
  const h16* Bb = BT + (long)blockIdx.z * sB;
  float*     Cb = C  + (long)blockIdx.z * sC;
  const int arow  = blockIdx.y * 16 + r;   // A fragment: lane -> M row
  const int bcol0 = blockIdx.x * 64 + r;   // B fragments: lane -> N col (4 tiles)
  v8f acc0 = {}, acc1 = {}, acc2 = {}, acc3 = {};
  for (int kk = 0; kk < K; kk += 32) {
    // A fragment: e -> K = half*16 + e (contiguous in A row)
    v16h a;
    const h16* ap = Ab + (long)arow * lda + kk + half * 16;
#pragma unroll
    for (int e = 0; e < 16; ++e) { a[e] = ap[e]; }
    // B fragments: e -> K = half*16 + e (contiguous in BT row)
    v16h b0, b1, b2, b3;
    const h16* bp0 = Bb + (long)(bcol0)      * ldbT + kk + half * 16;
    const h16* bp1 = Bb + (long)(bcol0 + 16) * ldbT + kk + half * 16;
    const h16* bp2 = Bb + (long)(bcol0 + 32) * ldbT + kk + half * 16;
    const h16* bp3 = Bb + (long)(bcol0 + 48) * ldbT + kk + half * 16;
#pragma unroll
    for (int e = 0; e < 16; ++e) { b0[e] = bp0[e]; b1[e] = bp1[e]; b2[e] = bp2[e]; b3[e] = bp3[e]; }
    acc0 = __builtin_amdgcn_wmma_f32_16x16x32_f16(false, a, false, b0, (short)0, acc0, false, false);
    acc1 = __builtin_amdgcn_wmma_f32_16x16x32_f16(false, a, false, b1, (short)0, acc1, false, false);
    acc2 = __builtin_amdgcn_wmma_f32_16x16x32_f16(false, a, false, b2, (short)0, acc2, false, false);
    acc3 = __builtin_amdgcn_wmma_f32_16x16x32_f16(false, a, false, b3, (short)0, acc3, false, false);
  }
  // C: lane -> N col = tile*16 + r ; VGPR v -> M row = half*8 + v
  float* cp = Cb + (long)(blockIdx.y * 16 + half * 8) * ldc + bcol0;
#pragma unroll
  for (int v = 0; v < 8; ++v) {
    cp[(long)v * ldc]      = acc0[v];
    cp[(long)v * ldc + 16] = acc1[v];
    cp[(long)v * ldc + 32] = acc2[v];
    cp[(long)v * ldc + 48] = acc3[v];
  }
}

// ---------------- channel L2 normalization ----------------
__global__ void k_norm(const float* __restrict__ out32, float* __restrict__ dout) {
  int p = blockIdx.x;     // 0..254
  int t = blockIdx.y;     // 0..127
  int c = threadIdx.x;    // 0..127
  float v = out32[t * 32768 + p * 128 + c];
  float s = v * v;
  for (int off = 16; off > 0; off >>= 1) s += __shfl_down(s, off, 32);
  __shared__ float sm[4];
  if ((c & 31) == 0) sm[c >> 5] = s;
  __syncthreads();
  float tot = sm[0] + sm[1] + sm[2] + sm[3];
  dout[(t * 255 + p) * 128 + c] = v / (sqrtf(tot) + 1e-6f);
}

// ---------------- workspace layout (bytes) ----------------
// FFT  @ 0        131072   (2 x 128 x 256 f16)
// A2   @ 131072   131072   (256 x 256 f16)
// LEGf @ 262144   2097152  (128 x 64 x 128 f16)
// LEGi @ 2359296  2097152  (128 x 128 x 64 f16)
// tc   @ 4456448  512
// R1   @ 4456960  8388608  : xhT -> FmBT -> flmA(+0)/yBT(+4194304) -> GBT
// R2   @ 12845568 8388608  : y32
// R3   @ 21234176 16777216 : Fm32 -> flm32(+0)+BbigT(+8388608) -> G32 -> out32
// total ~38 MB

extern "C" void kernel_launch(void* const* d_in, const int* in_sizes, int n_in,
                              void* d_out, int out_size, void* d_ws, size_t ws_size,
                              hipStream_t stream) {
  (void)in_sizes; (void)n_in; (void)out_size; (void)ws_size;
  const float* x      = (const float*)d_in[0];
  const float* t_emb  = (const float*)d_in[1];
  const float* ck     = (const float*)d_in[2];
  const float* cbias  = (const float*)d_in[3];
  const float* wre    = (const float*)d_in[4];
  const float* wim    = (const float*)d_in[5];
  const float* drw    = (const float*)d_in[6];
  const float* drb    = (const float*)d_in[7];
  const float* diw    = (const float*)d_in[8];
  const float* dib    = (const float*)d_in[9];
  float* out = (float*)d_out;

  char* ws = (char*)d_ws;
  h16*   FFT  = (h16*)(ws + 0);
  h16*   A2   = (h16*)(ws + 131072);
  h16*   LEGf = (h16*)(ws + 262144);
  h16*   LEGi = (h16*)(ws + 2359296);
  float* tc   = (float*)(ws + 4456448);
  char*  R1   = ws + 4456960;
  char*  R2   = ws + 12845568;
  char*  R3   = ws + 21234176;

  h16*   xhT   = (h16*)R1;                 // [t][c][p]   128*128*256
  float* Fm32  = (float*)R3;               // [t][2][128][128]
  h16*   FmBT  = (h16*)R1;                 // [m][k][t]   128*256*128
  float* flm32 = (float*)R3;               // [m][64][256]
  h16*   BbigT = (h16*)(R3 + 8388608);     // [l][n][k]   64*256*256
  h16*   flmA  = (h16*)R1;                 // [l][m][k]   64*128*256
  float* y32   = (float*)R2;               // [l][128][256]
  h16*   yBT   = (h16*)(R1 + 4194304);     // [m][n][l]   128*256*64
  float* G32   = (float*)R3;               // [m][128][256]
  h16*   GBT   = (h16*)R1;                 // [t][c][k]   128*128*256
  float* out32 = (float*)R3;               // [t][256][128]

  // ---- bases ----
  hipLaunchKernelGGL(k_basis,   dim3(128), dim3(128), 0, stream, LEGf, LEGi);
  hipLaunchKernelGGL(k_fourier, dim3(256), dim3(256), 0, stream, FFT, A2);
  hipLaunchKernelGGL(k_pack_xh, dim3(16384), dim3(256), 0, stream, x, xhT);
  hipLaunchKernelGGL(k_tc,      dim3(1), dim3(64), 0, stream, t_emb, drw, drb, diw, dib, tc);

  // ---- stage 1: forward Fourier (per-t GEMM, shared A, re+im) ----
  // M=128(mi) N=128(c) K=256(p)
  hipLaunchKernelGGL(k_gemm_wmma, dim3(2, 8, 128), dim3(32), 0, stream,
                     FFT,             xhT, Fm32,         256, 256, 256, 128,
                     0L, 32768L, 32768L);
  hipLaunchKernelGGL(k_gemm_wmma, dim3(2, 8, 128), dim3(32), 0, stream,
                     FFT + 128 * 256, xhT, Fm32 + 16384, 256, 256, 256, 128,
                     0L, 32768L, 32768L);
  hipLaunchKernelGGL(k_pack_FmB, dim3(16384), dim3(256), 0, stream, Fm32, FmBT);

  // ---- stage 2: forward Legendre (per-m GEMM)  M=64(l) N=256 K=128(t) ----
  hipLaunchKernelGGL(k_gemm_wmma, dim3(4, 4, 128), dim3(32), 0, stream,
                     LEGf, FmBT, flm32, 128, 128, 128, 256,
                     8192L, 32768L, 16384L);
  hipLaunchKernelGGL(k_pack_flmA, dim3(8192), dim3(256), 0, stream, flm32, flmA);

  // ---- stage 3: spectral mix + conv (per-l GEMM)  M=128(m) N=256 K=256 ----
  hipLaunchKernelGGL(k_build_bbig, dim3(4096), dim3(256), 0, stream, wre, wim, ck, tc, BbigT);
  hipLaunchKernelGGL(k_gemm_wmma, dim3(4, 8, 64), dim3(32), 0, stream,
                     flmA, BbigT, y32, 256, 256, 256, 256,
                     32768L, 65536L, 32768L);
  hipLaunchKernelGGL(k_pack_yB, dim3(8192), dim3(256), 0, stream, y32, cbias, yBT);

  // ---- stage 4: inverse Legendre (per-m GEMM)  M=128(t) N=256 K=64(l) ----
  hipLaunchKernelGGL(k_gemm_wmma, dim3(4, 8, 128), dim3(32), 0, stream,
                     LEGi, yBT, G32, 64, 64, 64, 256,
                     8192L, 16384L, 32768L);
  hipLaunchKernelGGL(k_pack_GB, dim3(16384), dim3(256), 0, stream, G32, GBT);

  // ---- stage 5: inverse Fourier -> real output (per-t GEMM, shared A)  M=256(p) N=128(c) K=256 ----
  hipLaunchKernelGGL(k_gemm_wmma, dim3(2, 16, 128), dim3(32), 0, stream,
                     A2, GBT, out32, 256, 256, 256, 128,
                     0L, 32768L, 32768L);

  // ---- channel normalization ----
  hipLaunchKernelGGL(k_norm, dim3(255, 128), dim3(128), 0, stream, out32, out);
}
